// FinalModel_26328149525076
// MI455X (gfx1250) — matrix operand order repaired and checked by reading, not compiled, not run on hardware
//
#include <hip/hip_runtime.h>
#include <math.h>
#include <stdint.h>

typedef __attribute__((ext_vector_type(16))) __bf16 v16bf;
typedef __attribute__((ext_vector_type(8)))  float  v8f;
typedef __attribute__((ext_vector_type(4)))  unsigned int u32x4;
typedef __attribute__((ext_vector_type(8)))  int i32x8;
typedef __attribute__((ext_vector_type(4)))  int i32x4;
typedef unsigned short u16;

#define DEVI __device__ __forceinline__

namespace {

constexpr int kB    = 1024;
constexpr int kT    = 91;
constexpr int kSEQ  = 274;
constexpr int kSEQP = 288;           // padded seq (multiple of 16)
constexpr float kPI = 3.14159265358979323846f;

union AF { u32x4 u[2]; v16bf v; };

DEVI u16 f2bf(float x){               // round-to-nearest-even f32 -> bf16
  unsigned u = __float_as_uint(x);
  return (u16)((u + 0x7fffu + ((u >> 16) & 1u)) >> 16);
}
DEVI float bflo(unsigned w){ return __uint_as_float(w << 16); }
DEVI float bfhi(unsigned w){ return __uint_as_float(w & 0xffff0000u); }

DEVI v8f wmma_bf16(v16bf a, v16bf b, v8f c){
  // D = A(16x32 bf16) * B(32x16 bf16) + C(16x16 f32)
  return __builtin_amdgcn_wmma_f32_16x16x32_bf16(false, a, false, b, (short)0, c, false, false);
}

// A fragment from LDS, row-major strip [16][pitch] bf16 (contiguous-k layout, same as frag_b):
// lane half h, row m=lane&15: elem j -> k=kb+h*16+j (contiguous, matches frag_b)
DEVI v16bf frag_a(const u16* As, int pitch, int lane, int kb){
  const int hf = (lane >> 4) & 1, m = lane & 15;
  const u16* p = As + m * pitch + kb + hf * 16;
  AF f;
  f.u[0] = *(const u32x4*)p;
  f.u[1] = *(const u32x4*)(p + 8);
  return f.v;
}
// B fragment: lane owns column n; 16 contiguous K values at p = &Bt[n][kb + half*16]
DEVI v16bf frag_b(const u16* p){
  AF f;
  f.u[0] = *(const u32x4*)p;
  f.u[1] = *(const u32x4*)(p + 8);
  return f.v;
}

// ---- Tensor Data Mover: DMA one 16x256 bf16 strip global -> LDS ----
// D# per ISA ch.8: 2D tile, data_size=2B, tile 256x16, row stride 256 elems.
// pad_enable with interval=128 dwords (2^(6+1)) and amount=4 dwords reproduces
// the bank-conflict-avoiding LDS pitch of 264 u16 (528B) per row.
DEVI void tdm_strip16x256(u16* As, const u16* src){
  const unsigned long long ga = (unsigned long long)(size_t)src;
  u32x4 g0;
  g0[0] = 1u;                                        // count=1 valid descriptor
  g0[1] = (unsigned)(size_t)(const void*)As;         // lds_addr (low 32 bits of AS3 addr)
  g0[2] = (unsigned)ga;                              // global_addr[31:0]
  g0[3] = (unsigned)((ga >> 32) & 0x01ffffffull)     // global_addr[56:32]
        | (2u << 30);                                // type=2 ("image")
  i32x8 g1;
  g1[0] = 0x07910000;        // data_size=2B | pad_enable | pad_interval=6 | pad_amount=3
  g1[1] = (int)(256u << 16); // tensor_dim0 = 256  (bits 63:48)
  g1[2] = (int)(16u  << 16); // tensor_dim0 hi = 0 ; tensor_dim1 = 16
  g1[3] = (int)(256u << 16); // tensor_dim1 hi = 0 ; tile_dim0 = 256
  g1[4] = 16;                // tile_dim1 = 16 ; tile_dim2 = 0
  g1[5] = 256;               // tensor_dim0_stride = 256 elems
  g1[6] = 0;
  g1[7] = 0;
  const i32x4 z4 = {0, 0, 0, 0};
  const i32x8 z8 = {0, 0, 0, 0, 0, 0, 0, 0};
  // 6-arg (clang-23 / therock-10.0) form: g0, g1, g2, g3, g4, cpol
  __builtin_amdgcn_tensor_load_to_lds(g0, g1, z4, z4, z8, 0);
}
// issue from one thread, wait TENSORcnt, then whole block syncs
DEVI void strip_to_lds(u16* As, const u16* src, int tid){
  if (tid == 0){
    tdm_strip16x256(As, src);
    __builtin_amdgcn_s_wait_tensorcnt(0);
  }
  __syncthreads();
}

// ---------------- weight transpose f32[256][256] -> bf16 WT[n][k] ----------------
__global__ void k_wt256(const float* __restrict__ W, u16* __restrict__ WT){
  int idx = blockIdx.x * 256 + threadIdx.x;       // 65536 elements
  int k = idx >> 8, n = idx & 255;
  WT[(size_t)n * 256 + k] = f2bf(W[idx]);
}

// ---------------- stream layer-1 MLP: relu(x[8] @ Wsa + bsa) -> bf16 ----------------
__global__ void k_embed_stream(const float* __restrict__ x0, const float* __restrict__ x1,
                               const float* __restrict__ x2,
                               const float* __restrict__ W0a, const float* __restrict__ b0a,
                               const float* __restrict__ W1a, const float* __restrict__ b1a,
                               const float* __restrict__ W2a, const float* __restrict__ b2a,
                               u16* __restrict__ midS){
  const int tokens = kB * kT;
  const int s   = blockIdx.x / tokens;
  const int rem = blockIdx.x - s * tokens;        // b*91 + t
  const float* x  = (s == 0 ? x0 : (s == 1 ? x1 : x2)) + (size_t)rem * 8;
  const float* W  = (s == 0 ? W0a : (s == 1 ? W1a : W2a));
  const float* bb = (s == 0 ? b0a : (s == 1 ? b1a : b2a));
  const int n = threadIdx.x;
  float acc = bb[n];
#pragma unroll
  for (int k2 = 0; k2 < 8; ++k2) acc += x[k2] * W[k2 * 256 + n];
  midS[((size_t)s * tokens + rem) * 256 + n] = f2bf(fmaxf(acc, 0.f));
}

// ---------------- context layer-1 MLP: relu(xcT[16] @ Wca + bca) ----------------
__global__ void k_embed_c(const float* __restrict__ xcT, const float* __restrict__ Wca,
                          const float* __restrict__ bca, u16* __restrict__ midC){
  const int b = blockIdx.x, n = threadIdx.x;
  const float* x = xcT + (size_t)b * 16;
  float acc = bca[n];
#pragma unroll
  for (int k2 = 0; k2 < 16; ++k2) acc += x[k2] * Wca[k2 * 256 + n];
  midC[(size_t)b * 256 + n] = f2bf(fmaxf(acc, 0.f));
}

// ---------------- layer-2 GEMM (+bias +pos) -> xIn (f32 + bf16) ----------------
template<int RPB, int ADDP>
__global__ void k_l2gemm(const u16* __restrict__ A, const u16* __restrict__ WT,
                         const float* __restrict__ bias, float* __restrict__ xinF,
                         u16* __restrict__ xinH, int tok_base){
  __shared__ __align__(16) u16 As[16 * 264];
  const int tid = threadIdx.x;
  const int r0  = blockIdx.x * 16;
  if (tid == 0) __builtin_prefetch(WT, 0, 1);
  strip_to_lds(As, A + (size_t)r0 * 256, tid);
  const int wave = tid >> 5, lane = tid & 31, hf = (lane >> 4) & 1;
  const int n0 = wave * 32 + (lane & 15), n1 = n0 + 16;
  v8f a0 = {}, a1 = {};
#pragma unroll
  for (int kb = 0; kb < 256; kb += 32){
    v16bf a = frag_a(As, 264, lane, kb);
    a0 = wmma_bf16(a, frag_b(WT + (size_t)n0 * 256 + kb + hf * 16), a0);
    a1 = wmma_bf16(a, frag_b(WT + (size_t)n1 * 256 + kb + hf * 16), a1);
  }
#pragma unroll
  for (int r = 0; r < 8; ++r){
    const int gr  = r0 + r + hf * 8;
    const int b   = gr / RPB, t = gr - b * RPB;   // RPB is compile-time: mul/shift
    const int tok = tok_base + t;
    const size_t orow = ((size_t)b * kSEQP + tok) * 256;
#pragma unroll
    for (int h2 = 0; h2 < 2; ++h2){
      const int n = h2 ? n1 : n0;
      float v = (h2 ? a1[r] : a0[r]) + bias[n];
      if (ADDP){
        float ang = (float)(t - 45) / (float)((n >> 1) + 1) * kPI;
        v += (n & 1) ? __cosf(ang) : __sinf(ang);
      }
      xinF[orow + n] = v;
      xinH[orow + n] = f2bf(v);
    }
  }
}

// ---------------- QKV GEMM: q,k row-major bf16; v transposed [B][256][288] ----------------
__global__ void k_qkv(const u16* __restrict__ xinH,
                      const u16* __restrict__ WTq, const u16* __restrict__ WTk,
                      const u16* __restrict__ WTv,
                      u16* __restrict__ qH, u16* __restrict__ kH, u16* __restrict__ vT){
  __shared__ __align__(16) u16 As[16 * 264];
  const int tid = threadIdx.x, r0 = blockIdx.x * 16;
  strip_to_lds(As, xinH + (size_t)r0 * 256, tid);
  const int wave = tid >> 5, lane = tid & 31, hf = (lane >> 4) & 1;
  const int n0 = wave * 32 + (lane & 15), n1 = n0 + 16;
  v8f q0 = {}, q1 = {}, k0 = {}, k1 = {}, v0 = {}, v1 = {};
#pragma unroll
  for (int kb = 0; kb < 256; kb += 32){
    v16bf a = frag_a(As, 264, lane, kb);
    const size_t o0 = (size_t)n0 * 256 + kb + hf * 16;
    const size_t o1 = (size_t)n1 * 256 + kb + hf * 16;
    q0 = wmma_bf16(a, frag_b(WTq + o0), q0);
    q1 = wmma_bf16(a, frag_b(WTq + o1), q1);
    k0 = wmma_bf16(a, frag_b(WTk + o0), k0);
    k1 = wmma_bf16(a, frag_b(WTk + o1), k1);
    v0 = wmma_bf16(a, frag_b(WTv + o0), v0);
    v1 = wmma_bf16(a, frag_b(WTv + o1), v1);
  }
#pragma unroll
  for (int r = 0; r < 8; ++r){
    const int gr = r0 + r + hf * 8;
    const int b = gr / kSEQP, trow = gr - b * kSEQP;
    const size_t orow = (size_t)gr * 256;
    qH[orow + n0] = f2bf(q0[r]);  qH[orow + n1] = f2bf(q1[r]);
    kH[orow + n0] = f2bf(k0[r]);  kH[orow + n1] = f2bf(k1[r]);
    vT[((size_t)b * 256 + n0) * kSEQP + trow] = f2bf(v0[r]);
    vT[((size_t)b * 256 + n1) * kSEQP + trow] = f2bf(v1[r]);
  }
}

// ---------------- attention: scores (WMMA) -> masked softmax -> attn@v (WMMA) ----------------
__global__ void k_attn(const u16* __restrict__ qH, const u16* __restrict__ kH,
                       const u16* __restrict__ vT, const int* __restrict__ mask,
                       u16* __restrict__ avH){
  __shared__ __align__(16) u16   qs[16 * 264];
  __shared__ __align__(16) float sc[16 * 296];
  __shared__ __align__(16) u16   at[16 * 296];
  const int tid = threadIdx.x;
  const int qt  = blockIdx.x;                 // 0..17 (query tile)
  const int b   = blockIdx.y;                 // batch
  const int wave = tid >> 5, lane = tid & 31, hf = (lane >> 4) & 1;
  strip_to_lds(qs, qH + ((size_t)b * kSEQP + qt * 16) * 256, tid);
  const float scale = rsqrtf((float)kSEQ);    // reference divides by sqrt(SEQ)
  for (int kt = wave; kt < 18; kt += 8){
    v8f acc = {};
    const int j = kt * 16 + (lane & 15);
    const u16* kp = kH + ((size_t)b * kSEQP + j) * 256 + hf * 16;
#pragma unroll
    for (int kb = 0; kb < 256; kb += 32)
      acc = wmma_bf16(frag_a(qs, 264, lane, kb), frag_b(kp + kb), acc);
    const bool ok = (j < kSEQ) && (mask[(size_t)b * kSEQ + j] != 0);
#pragma unroll
    for (int r = 0; r < 8; ++r)
      sc[(r + hf * 8) * 296 + j] = ok ? acc[r] * scale : -1e30f;
  }
  __syncthreads();
  { // softmax: 16 threads per row, shfl reductions stay inside 16-lane groups
    const int row = tid >> 4, sub = tid & 15;
    float mx = -3.4e38f;
    for (int c = sub; c < 288; c += 16) mx = fmaxf(mx, sc[row * 296 + c]);
    for (int o = 8; o; o >>= 1) mx = fmaxf(mx, __shfl_xor(mx, o, 32));
    float sum = 0.f;
    for (int c = sub; c < 288; c += 16){
      float e = __expf(sc[row * 296 + c] - mx);
      sc[row * 296 + c] = e;
      sum += e;
    }
    for (int o = 8; o; o >>= 1) sum += __shfl_xor(sum, o, 32);
    const float inv = 1.f / sum;
    for (int c = sub; c < 288; c += 16) at[row * 296 + c] = f2bf(sc[row * 296 + c] * inv);
  }
  __syncthreads();
  { // o = attn @ v : A from LDS (K = 288 keys), B from vT (d contiguous over keys)
    const int n0 = wave * 32 + (lane & 15), n1 = n0 + 16;
    v8f o0 = {}, o1 = {};
    const u16* v0p = vT + ((size_t)b * 256 + n0) * kSEQP + hf * 16;
    const u16* v1p = vT + ((size_t)b * 256 + n1) * kSEQP + hf * 16;
#pragma unroll
    for (int kb = 0; kb < 288; kb += 32){
      v16bf a = frag_a(at, 296, lane, kb);
      o0 = wmma_bf16(a, frag_b(v0p + kb), o0);
      o1 = wmma_bf16(a, frag_b(v1p + kb), o1);
    }
#pragma unroll
    for (int r = 0; r < 8; ++r){
      const size_t orow = ((size_t)b * kSEQP + qt * 16 + r + hf * 8) * 256;
      avH[orow + n0] = f2bf(o0[r]);
      avH[orow + n1] = f2bf(o1[r]);
    }
  }
}

// ---------------- GEMM + bias + relu -> bf16 ----------------
__global__ void k_gemm_relu(const u16* __restrict__ A, const u16* __restrict__ WT,
                            const float* __restrict__ bias, u16* __restrict__ out){
  __shared__ __align__(16) u16 As[16 * 264];
  const int tid = threadIdx.x, r0 = blockIdx.x * 16;
  if (tid == 0) __builtin_prefetch(WT, 0, 1);
  strip_to_lds(As, A + (size_t)r0 * 256, tid);
  const int wave = tid >> 5, lane = tid & 31, hf = (lane >> 4) & 1;
  const int n0 = wave * 32 + (lane & 15), n1 = n0 + 16;
  v8f a0 = {}, a1 = {};
#pragma unroll
  for (int kb = 0; kb < 256; kb += 32){
    v16bf a = frag_a(As, 264, lane, kb);
    a0 = wmma_bf16(a, frag_b(WT + (size_t)n0 * 256 + kb + hf * 16), a0);
    a1 = wmma_bf16(a, frag_b(WT + (size_t)n1 * 256 + kb + hf * 16), a1);
  }
#pragma unroll
  for (int r = 0; r < 8; ++r){
    const size_t orow = (size_t)(r0 + r + hf * 8) * 256;
    out[orow + n0] = f2bf(fmaxf(a0[r] + bias[n0], 0.f));
    out[orow + n1] = f2bf(fmaxf(a1[r] + bias[n1], 0.f));
  }
}

// ---------------- GEMM (+bias) + residual(xIn f32) + LayerNorm -> bf16 ----------------
__global__ void k_gemm_ln(const u16* __restrict__ A, const u16* __restrict__ WT,
                          const float* __restrict__ bias, const float* __restrict__ xinF,
                          const float* __restrict__ gam, const float* __restrict__ bet,
                          u16* __restrict__ out){
  __shared__ __align__(16) u16 As[16 * 264];
  __shared__ float Cs[16 * 260];
  const int tid = threadIdx.x, r0 = blockIdx.x * 16;
  strip_to_lds(As, A + (size_t)r0 * 256, tid);
  const int wave = tid >> 5, lane = tid & 31, hf = (lane >> 4) & 1;
  const int n0 = wave * 32 + (lane & 15), n1 = n0 + 16;
  v8f a0 = {}, a1 = {};
#pragma unroll
  for (int kb = 0; kb < 256; kb += 32){
    v16bf a = frag_a(As, 264, lane, kb);
    a0 = wmma_bf16(a, frag_b(WT + (size_t)n0 * 256 + kb + hf * 16), a0);
    a1 = wmma_bf16(a, frag_b(WT + (size_t)n1 * 256 + kb + hf * 16), a1);
  }
  const float bi0 = bias ? bias[n0] : 0.f;
  const float bi1 = bias ? bias[n1] : 0.f;
#pragma unroll
  for (int r = 0; r < 8; ++r){
    const int lm = r + hf * 8;
    Cs[lm * 260 + n0] = a0[r] + bi0;
    Cs[lm * 260 + n1] = a1[r] + bi1;
  }
  __syncthreads();
  const int row = tid >> 4, sub = tid & 15;
  const int gr = r0 + row;
  float v[16], s = 0.f;
#pragma unroll
  for (int i = 0; i < 16; ++i){
    const int c = sub * 16 + i;
    const float val = Cs[row * 260 + c] + xinF[(size_t)gr * 256 + c];
    v[i] = val; s += val;
  }
  for (int o = 8; o; o >>= 1) s += __shfl_xor(s, o, 32);
  const float mean = s * (1.f / 256.f);
  float vs = 0.f;
#pragma unroll
  for (int i = 0; i < 16; ++i){ const float d = v[i] - mean; vs += d * d; }
  for (int o = 8; o; o >>= 1) vs += __shfl_xor(vs, o, 32);
  const float inv = rsqrtf(vs * (1.f / 256.f) + 1e-5f);
#pragma unroll
  for (int i = 0; i < 16; ++i){
    const int c = sub * 16 + i;
    out[(size_t)gr * 256 + c] = f2bf((v[i] - mean) * inv * gam[c] + bet[c]);
  }
}

// ---------------- head: per-token scalar + ragged segment-mean pooling ----------------
__global__ void k_head(const u16* __restrict__ gH, const float* __restrict__ Wg2,
                       const float* __restrict__ bg2, float* __restrict__ outv){
  __shared__ float tok[288];
  __shared__ float part[3];
  const int b = blockIdx.x, tid = threadIdx.x;
  for (int t = tid; t < 288; t += 256) tok[t] = 0.f;
  __syncthreads();
  for (int t = tid; t < kSEQ; t += 256){
    const u16* gp = gH + ((size_t)b * kSEQP + t) * 256;
    float d = 0.f;
    for (int k2 = 0; k2 < 256; k2 += 8){
      const u32x4 u = *(const u32x4*)(gp + k2);
      d += bflo(u.x) * Wg2[k2 + 0] + bfhi(u.x) * Wg2[k2 + 1];
      d += bflo(u.y) * Wg2[k2 + 2] + bfhi(u.y) * Wg2[k2 + 3];
      d += bflo(u.z) * Wg2[k2 + 4] + bfhi(u.z) * Wg2[k2 + 5];
      d += bflo(u.w) * Wg2[k2 + 6] + bfhi(u.w) * Wg2[k2 + 7];
    }
    tok[t] = d + bg2[0];
  }
  __syncthreads();
  const int wave = tid >> 5, lane = tid & 31;
  if (wave < 3){
    const int base = wave * 91;
    const int len  = (wave == 2) ? 92 : 91;
    float s = 0.f;
    for (int i = lane; i < len; i += 32) s += tok[base + i];
    for (int o = 16; o; o >>= 1) s += __shfl_xor(s, o, 32);
    if (lane == 0) part[wave] = s / (float)len;
  }
  __syncthreads();
  if (tid == 0) outv[b] = part[0] + part[1] + part[2];
}

} // namespace

extern "C" void kernel_launch(void* const* d_in, const int* in_sizes, int n_in,
                              void* d_out, int out_size, void* d_ws, size_t ws_size,
                              hipStream_t stream){
  (void)in_sizes; (void)n_in; (void)out_size; (void)ws_size;
  // input dict order
  const float* x0  = (const float*)d_in[0];
  const float* x1  = (const float*)d_in[1];
  const float* x2  = (const float*)d_in[2];
  const float* xcT = (const float*)d_in[3];
  const int*   msk = (const int*)  d_in[4];

  char* base = (char*)d_ws;
  size_t off = 0;
  auto take = [&](size_t bytes)->char*{
    char* p = base + off;
    off += (bytes + 255) & ~(size_t)255;
    return p;
  };
  const size_t rows = (size_t)kB * kSEQP;                 // 294912 padded rows
  float* xinF = (float*)take(rows * 256 * 4);
  u16*   xinH = (u16*)  take(rows * 256 * 2);
  u16*   qH   = (u16*)  take(rows * 256 * 2);
  u16*   kHb  = (u16*)  take(rows * 256 * 2);
  u16*   vTb  = (u16*)  take(rows * 256 * 2);             // [B][256][288]
  u16*   avH  = (u16*)  take(rows * 256 * 2);
  u16*   y1H  = (u16*)  take(rows * 256 * 2);
  u16*   m1H  = (u16*)  take(rows * 256 * 2);
  u16*   midS = (u16*)  take((size_t)3 * kB * kT * 256 * 2);
  u16*   midC = (u16*)  take((size_t)kB * 256 * 2);
  u16*   wtB  = (u16*)  take((size_t)11 * 65536 * 2);
  u16*   y2H  = qH;                                       // reuse after attention
  u16*   gH   = kHb;                                      // reuse after attention

  // transpose weights -> bf16 [n][k]:
  // 0:W0b 1:W1b 2:W2b 3:Wcb 4:Wq 5:Wk 6:Wv 7:Wo 8:Wf1 9:Wf2 10:Wg1
  const int wsrc[11] = {7, 11, 15, 19, 21, 22, 23, 24, 29, 31, 33};
  for (int i = 0; i < 11; ++i)
    k_wt256<<<256, 256, 0, stream>>>((const float*)d_in[wsrc[i]], wtB + (size_t)i * 65536);
  auto WT = [&](int i){ return (const u16*)(wtB + (size_t)i * 65536); };

  // layer-1 MLPs
  k_embed_stream<<<3 * kB * kT, 256, 0, stream>>>(x0, x1, x2,
      (const float*)d_in[5],  (const float*)d_in[6],
      (const float*)d_in[9],  (const float*)d_in[10],
      (const float*)d_in[13], (const float*)d_in[14], midS);
  k_embed_c<<<kB, 256, 0, stream>>>(xcT, (const float*)d_in[17], (const float*)d_in[18], midC);

  // layer-2 GEMMs -> xIn (f32 + bf16), + positional encoding for streams
  const int bsb[3] = {8, 12, 16};
  for (int s = 0; s < 3; ++s)
    k_l2gemm<kT, 1><<<(kB * kT) / 16, 256, 0, stream>>>(midS + (size_t)s * kB * kT * 256,
        WT(s), (const float*)d_in[bsb[s]], xinF, xinH, s * kT);
  k_l2gemm<1, 0><<<kB / 16, 256, 0, stream>>>(midC, WT(3), (const float*)d_in[20],
                                              xinF, xinH, 273);

  // QKV
  k_qkv<<<(int)(rows / 16), 256, 0, stream>>>(xinH, WT(4), WT(5), WT(6), qH, kHb, vTb);

  // attention
  k_attn<<<dim3(18, kB), 256, 0, stream>>>(qH, kHb, vTb, msk, avH);

  // o @ Wo + xIn -> LN1 -> y1
  k_gemm_ln<<<(int)(rows / 16), 256, 0, stream>>>(avH, WT(7), nullptr, xinF,
      (const float*)d_in[25], (const float*)d_in[26], y1H);
  // ffn mid = relu(y1 @ Wf1 + bf1)
  k_gemm_relu<<<(int)(rows / 16), 256, 0, stream>>>(y1H, WT(8), (const float*)d_in[30], m1H);
  // y2 = LN2(mid @ Wf2 + bf2 + xIn)
  k_gemm_ln<<<(int)(rows / 16), 256, 0, stream>>>(m1H, WT(9), (const float*)d_in[32], xinF,
      (const float*)d_in[27], (const float*)d_in[28], y2H);
  // g = relu(y2 @ Wg1 + bg1)
  k_gemm_relu<<<(int)(rows / 16), 256, 0, stream>>>(y2H, WT(10), (const float*)d_in[34], gH);
  // head + pooling
  k_head<<<kB, 256, 0, stream>>>(gH, (const float*)d_in[35], (const float*)d_in[36],
                                 (float*)d_out);
}
